// Attention_11424613007648
// MI455X (gfx1250) — matrix operand order repaired and checked by reading, not compiled, not run on hardware
//
#include <hip/hip_runtime.h>
#include <hip/hip_bf16.h>

#define B_   2
#define D_   2048
#define H_   32
#define DH_  64
#define LQ_  1024
#define LK_  1024

typedef __attribute__((ext_vector_type(16))) __bf16          v16bf;
typedef __attribute__((ext_vector_type(8)))  float           v8f;
typedef __attribute__((ext_vector_type(8)))  unsigned short  u16x8;
typedef __attribute__((ext_vector_type(16))) unsigned short  u16x16;

// ---- helpers -------------------------------------------------------------

__device__ __forceinline__ unsigned short f2bf(float f) {
  unsigned int u = __builtin_bit_cast(unsigned int, f);
  u = (u + 0x7FFFu + ((u >> 16) & 1u)) >> 16;   // round-to-nearest-even
  return (unsigned short)u;
}

// A-operand fragment (16x32 bf16, MxK): lane holds row m = lane&15.
// Halves 0..15 -> k = hi*16 + 0..15 contiguous.
// `row` must already point at A[m-row][k-chunk base]; hi = lane>>4.
__device__ __forceinline__ v16bf load_a_frag(const unsigned short* row, int hi) {
  u16x8 lo = *(const u16x8*)(row + hi * 16);
  u16x8 hp = *(const u16x8*)(row + hi * 16 + 8);
  u16x16 u = __builtin_shufflevector(lo, hp, 0,1,2,3,4,5,6,7,8,9,10,11,12,13,14,15);
  return __builtin_bit_cast(v16bf, u);
}

// B-operand fragment (32x16 bf16, KxN): lane holds col n = lane&15,
// k = hi*16 + 0..15 contiguous. `p` points at BT[n-row][k-chunk + hi*16].
__device__ __forceinline__ v16bf load_b_frag(const unsigned short* p) {
  return __builtin_bit_cast(v16bf, *(const u16x16*)p);
}

__device__ __forceinline__ v8f wmma_bf16(v16bf a, v16bf b, v8f c) {
  return __builtin_amdgcn_wmma_f32_16x16x32_bf16(false, a, false, b, (short)0, c,
                                                 false, false);
}

// ---- conversion kernels --------------------------------------------------

__global__ void f32_to_bf16_kernel(const float* __restrict__ src,
                                   unsigned short* __restrict__ dst, int n) {
  int i = blockIdx.x * blockDim.x + threadIdx.x;
  if (i < n) dst[i] = f2bf(src[i]);
}

// src: [b][Dm][L] f32  ->  dst: [b][L][Dm] bf16   (transpose for B-operand)
__global__ void transpose_f32_to_bf16_kernel(const float* __restrict__ src,
                                             unsigned short* __restrict__ dst,
                                             int Dm, int L, int total) {
  int i = blockIdx.x * blockDim.x + threadIdx.x;
  if (i >= total) return;
  int l = i % L;
  int r = i / L;
  int d = r % Dm;
  int b = r / Dm;
  dst[((size_t)b * L + l) * Dm + d] = f2bf(src[i]);
}

// ---- register-blocked bf16 WMMA GEMM -------------------------------------
// C[b][M][N] = A[M][K] (row-major bf16, shared across batch)
//            x BT[b][N][K] (each output column stored as a contiguous K-row)
// Per-wave output block: 64 (m) x 32 (n) = 4x2 WMMA tiles; per 32-k step we
// load 4 A-frags + 2 B-frags (6 KB/wave) and issue 8 WMMAs (131 KFLOP)
// -> ~22 FLOP/B operand intensity, 2.7x better than a single-tile wave.
// mode 0: out bf16 natural   [b][M][N]
// mode 1: out bf16 head-T    [b][o>>6][N][o&63]   (o = output row)
// mode 2: out f32  natural   [b][M][N]
__global__ void gemm_bf16_wmma(const unsigned short* __restrict__ A,
                               const unsigned short* __restrict__ BT,
                               void* __restrict__ out,
                               int M, int N, int K, int mode) {
  const int lane = threadIdx.x & 31;
  const int wave = threadIdx.x >> 5;
  const int b = blockIdx.y;
  const int nsup = N >> 5;                       // 32-wide n supertiles
  const int tile = blockIdx.x * (blockDim.x >> 5) + wave;
  const int mt = tile / nsup;                    // 64-row m supertile
  const int nt = tile % nsup;
  if (mt >= (M >> 6)) return;                    // wave-uniform

  const int m  = lane & 15;
  const int hi = lane >> 4;

  const unsigned short* Arow0 = A + (size_t)(mt * 64 + m) * K;
  const unsigned short* Brow0 = BT + (size_t)b * N * K + (size_t)(nt * 32 + m) * K;

  v8f c[4][2];
#pragma unroll
  for (int i = 0; i < 4; ++i)
#pragma unroll
    for (int j = 0; j < 2; ++j) c[i][j] = v8f{};

  for (int kk = 0; kk < K; kk += 32) {
    v16bf af[4];
#pragma unroll
    for (int i = 0; i < 4; ++i)
      af[i] = load_a_frag(Arow0 + (size_t)i * 16 * K + kk, hi);
    v16bf bf[2];
#pragma unroll
    for (int j = 0; j < 2; ++j)
      bf[j] = load_b_frag(Brow0 + (size_t)j * 16 * K + kk + hi * 16);
#pragma unroll
    for (int i = 0; i < 4; ++i)
#pragma unroll
      for (int j = 0; j < 2; ++j)
        c[i][j] = wmma_bf16(af[i], bf[j], c[i][j]);
  }

  if (mode == 0) {
    unsigned short* O = (unsigned short*)out + (size_t)b * M * N;
#pragma unroll
    for (int i = 0; i < 4; ++i)
#pragma unroll
      for (int j = 0; j < 2; ++j)
#pragma unroll
        for (int r = 0; r < 8; ++r) {
          int o  = mt * 64 + i * 16 + r + hi * 8;
          int nq = nt * 32 + j * 16 + m;
          O[(size_t)o * N + nq] = f2bf(c[i][j][r]);
        }
  } else if (mode == 1) {
    unsigned short* O = (unsigned short*)out;
#pragma unroll
    for (int i = 0; i < 4; ++i)
#pragma unroll
      for (int j = 0; j < 2; ++j)
#pragma unroll
        for (int r = 0; r < 8; ++r) {
          int o  = mt * 64 + i * 16 + r + hi * 8;
          int nq = nt * 32 + j * 16 + m;
          O[(((size_t)b * H_ + (o >> 6)) * N + nq) * DH_ + (o & 63)] = f2bf(c[i][j][r]);
        }
  } else {
    float* O = (float*)out + (size_t)b * M * N;
#pragma unroll
    for (int i = 0; i < 4; ++i)
#pragma unroll
      for (int j = 0; j < 2; ++j)
#pragma unroll
        for (int r = 0; r < 8; ++r) {
          int o  = mt * 64 + i * 16 + r + hi * 8;
          int nq = nt * 32 + j * 16 + m;
          O[(size_t)o * N + nq] = c[i][j][r];
        }
  }
}

// ---- fused flash attention ----------------------------------------------
// hqT: [b][h][q][64] bf16, hkT: [b][h][k][64] bf16, hv: [b][o][LK] bf16
// pb:  [h][LK][LQ] f32, mask: [b][LK][LQ] i32
// ctxT out: [b][q][o] bf16   (B-operand layout for the final GEMM)
__global__ void flash_attn_wmma(const unsigned short* __restrict__ hqT,
                                const unsigned short* __restrict__ hkT,
                                const unsigned short* __restrict__ hv,
                                const float* __restrict__ pb,
                                const int* __restrict__ mask,
                                unsigned short* __restrict__ ctxT) {
  const int lane = threadIdx.x & 31;
  const int wave = threadIdx.x >> 5;
  const int QT = LQ_ / 16;                       // 64 q-tiles
  const int t = blockIdx.x * (blockDim.x >> 5) + wave;
  const int b   = t / (H_ * QT);
  const int rem = t % (H_ * QT);
  const int h   = rem / QT;
  const int qb  = (rem % QT) * 16;

  const int m  = lane & 15;
  const int hi = lane >> 4;
  const int qg = qb + m;

  // Q fragments (B-operand, contraction dim = d), loop-invariant
  const unsigned short* qrow = hqT + (((size_t)b * H_ + h) * LQ_ + qg) * DH_;
  v16bf bq0 = load_b_frag(qrow + 0  + hi * 16);
  v16bf bq1 = load_b_frag(qrow + 32 + hi * 16);

  const unsigned short* kbase = hkT + ((size_t)b * H_ + h) * LK_ * DH_;
  const unsigned short* vbase = hv + ((size_t)b * D_ + h * DH_) * LK_;

  v8f ctx[4] = {v8f{}, v8f{}, v8f{}, v8f{}};
  float mrun = -1e30f, lrun = 0.f;

  for (int kb = 0; kb < LK_; kb += 32) {
    float p0[8], p1[8];
#pragma unroll
    for (int sub = 0; sub < 2; ++sub) {
      const unsigned short* arow = kbase + (size_t)(kb + sub * 16 + m) * DH_;
      v8f s = {};
      s = wmma_bf16(load_a_frag(arow + 0,  hi), bq0, s);
      s = wmma_bf16(load_a_frag(arow + 32, hi), bq1, s);
      float* pp = sub ? p1 : p0;
#pragma unroll
      for (int r = 0; r < 8; ++r) {
        int kg = kb + sub * 16 + r + hi * 8;
        float sv = s[r] + pb[((size_t)h * LK_ + kg) * LQ_ + qg];
        if (mask[((size_t)b * LK_ + kg) * LQ_ + qg] == 0) sv = -1e30f;
        pp[r] = sv;
      }
    }

    // running softmax statistics over the k axis (column-wise in the tile;
    // lane l and lane l^16 hold the same column -> shfl_xor(16) completes it)
    float tmax = p0[0];
#pragma unroll
    for (int r = 0; r < 8; ++r) tmax = fmaxf(tmax, fmaxf(p0[r], p1[r]));
    tmax = fmaxf(tmax, __shfl_xor(tmax, 16, 32));
    float mnew  = fmaxf(mrun, tmax);
    float alpha = __expf(mrun - mnew);
    float psum = 0.f;
#pragma unroll
    for (int r = 0; r < 8; ++r) {
      p0[r] = __expf(p0[r] - mnew);
      p1[r] = __expf(p1[r] - mnew);
      psum += p0[r] + p1[r];
    }
    psum += __shfl_xor(psum, 16, 32);
    lrun = lrun * alpha + psum;
    mrun = mnew;
#pragma unroll
    for (int dt = 0; dt < 4; ++dt)
#pragma unroll
      for (int r = 0; r < 8; ++r) ctx[dt][r] *= alpha;

    // convert P (two C-layout subtiles) into one 32x16 B-operand fragment
    u16x16 pu = {};
#pragma unroll
    for (int j = 0; j < 8; ++j) {
      float x0 = __shfl_xor(p0[j], 16, 32);
      float x1 = __shfl_xor(p1[j], 16, 32);
      float vlo = hi ? x1    : p0[j];   // element j   (k = hi*16 + j)
      float vhi = hi ? p1[j] : x0;      // element j+8 (k = hi*16 + 8 + j)
      pu[j]     = f2bf(vlo);
      pu[j + 8] = f2bf(vhi);
    }
    v16bf pf = __builtin_bit_cast(v16bf, pu);

    // ctx[d, q] += V(16d x 32k) x P(32k x 16q), 4 d-tiles cover DH=64
#pragma unroll
    for (int dt = 0; dt < 4; ++dt) {
      const unsigned short* vrow = vbase + (size_t)(dt * 16 + m) * LK_ + kb;
      ctx[dt] = wmma_bf16(load_a_frag(vrow, hi), pf, ctx[dt]);
    }
  }

  float inv = (lrun > 0.f) ? (1.0f / lrun) : 0.f;
#pragma unroll
  for (int dt = 0; dt < 4; ++dt)
#pragma unroll
    for (int r = 0; r < 8; ++r) {
      int o = h * DH_ + dt * 16 + r + hi * 8;
      ctxT[((size_t)b * LQ_ + qg) * D_ + o] = f2bf(ctx[dt][r] * inv);
    }
}

// ---- launcher ------------------------------------------------------------

extern "C" void kernel_launch(void* const* d_in, const int* in_sizes, int n_in,
                              void* d_out, int out_size, void* d_ws, size_t ws_size,
                              hipStream_t stream) {
  const float* hidden_q  = (const float*)d_in[0];
  const float* hidden_kv = (const float*)d_in[1];
  const int*   mask      = (const int*)d_in[2];
  const float* pbias     = (const float*)d_in[3];
  const float* Wq        = (const float*)d_in[4];
  const float* Wk        = (const float*)d_in[5];
  const float* Wv        = (const float*)d_in[6];
  const float* Wo        = (const float*)d_in[7];

  const size_t WB   = (size_t)D_ * D_ * sizeof(unsigned short);       // 8 MiB
  const size_t HB   = (size_t)B_ * LQ_ * D_ * sizeof(unsigned short); // 8 MiB
  char* ws = (char*)d_ws;
  unsigned short* WqB   = (unsigned short*)(ws);
  unsigned short* WkB   = (unsigned short*)(ws + WB);
  unsigned short* WvB   = (unsigned short*)(ws + 2 * WB);
  unsigned short* WoB   = (unsigned short*)(ws + 3 * WB);
  unsigned short* hqIn  = (unsigned short*)(ws + 4 * WB);
  unsigned short* hkvIn = (unsigned short*)(ws + 4 * WB + HB);
  unsigned short* hqT   = (unsigned short*)(ws + 4 * WB + 2 * HB);
  unsigned short* hkT   = (unsigned short*)(ws + 4 * WB + 3 * HB);
  unsigned short* hvN   = (unsigned short*)(ws + 4 * WB + 4 * HB);
  unsigned short* ctxT  = (unsigned short*)(ws + 4 * WB + 5 * HB);

  const int wn = D_ * D_;                       // 4 Mi elements
  f32_to_bf16_kernel<<<(wn + 255) / 256, 256, 0, stream>>>(Wq, WqB, wn);
  f32_to_bf16_kernel<<<(wn + 255) / 256, 256, 0, stream>>>(Wk, WkB, wn);
  f32_to_bf16_kernel<<<(wn + 255) / 256, 256, 0, stream>>>(Wv, WvB, wn);
  f32_to_bf16_kernel<<<(wn + 255) / 256, 256, 0, stream>>>(Wo, WoB, wn);

  const int hn = B_ * D_ * LQ_;
  transpose_f32_to_bf16_kernel<<<(hn + 255) / 256, 256, 0, stream>>>(
      hidden_q, hqIn, D_, LQ_, hn);
  transpose_f32_to_bf16_kernel<<<(hn + 255) / 256, 256, 0, stream>>>(
      hidden_kv, hkvIn, D_, LK_, hn);

  // M=2048, N=1024: (2048/64)*(1024/32) = 1024 waves/batch, 8 waves/block
  dim3 g(128, B_);
  gemm_bf16_wmma<<<g, 256, 0, stream>>>(WqB, hqIn,  hqT,  D_, LQ_, D_, 1);
  gemm_bf16_wmma<<<g, 256, 0, stream>>>(WkB, hkvIn, hkT,  D_, LK_, D_, 1);
  gemm_bf16_wmma<<<g, 256, 0, stream>>>(WvB, hkvIn, hvN,  D_, LK_, D_, 0);

  // 2*32*64 = 4096 waves, 8 waves per block
  flash_attn_wmma<<<512, 256, 0, stream>>>(hqT, hkT, hvN, pbias, mask, ctxT);

  gemm_bf16_wmma<<<g, 256, 0, stream>>>(WoB, ctxT, d_out, D_, LQ_, D_, 2);
}
